// SelfAttention_8942121910998
// MI455X (gfx1250) — matrix operand order repaired and checked by reading, not compiled, not run on hardware
//
#include <hip/hip_runtime.h>

#define B_ 8
#define T_ 2048
#define C_ 1024
#define D_ 128

typedef __attribute__((ext_vector_type(16))) _Float16 v16h;
typedef __attribute__((ext_vector_type(8)))  float    v8f;
typedef __attribute__((ext_vector_type(4)))  unsigned int u32x4;
typedef __attribute__((ext_vector_type(8)))  int          i32x8;
typedef __attribute__((ext_vector_type(4)))  int          i32x4;

union V16 { uint4 u[2]; v16h v; };

#if __has_builtin(__builtin_amdgcn_tensor_load_to_lds) && __has_builtin(__builtin_amdgcn_s_wait_tensorcnt)
#define HAVE_TDM 1
#else
#define HAVE_TDM 0
#endif

// ---------------------------------------------------------------------------
// Kernel 0: Wt[mat][d][c] = (f16) W[mat][c][d]  (transpose + convert once).
// ---------------------------------------------------------------------------
__global__ __launch_bounds__(256) void wt_convert_kernel(
    const float* __restrict__ Wq,
    const float* __restrict__ Wk,
    const float* __restrict__ Wv,
    _Float16* __restrict__ Wt)   // [3][D][C] f16
{
  const int idx = blockIdx.x * 256 + threadIdx.x;     // 3*C*D total
  const int mat = idx / (C_ * D_);
  const int r   = idx % (C_ * D_);
  const int c   = r / D_;                             // coalesced read along d
  const int d   = r % D_;
  const float* W = (mat == 0) ? Wq : (mat == 1) ? Wk : Wv;
  Wt[(size_t)mat * C_ * D_ + (size_t)d * C_ + c] = (_Float16)W[(size_t)c * D_ + d];
}

// ---------------------------------------------------------------------------
// Kernel 1: q/k/v = x @ W. 8 waves/block = 128 rows; Wt chunks (32x128 f16)
// double-buffered through LDS; next chunk's global loads issued before the
// current chunk's 8 WMMAs. q,k stored row-major f16; v stored TRANSPOSED
// (vT[b][d][t]) so kernel 2 can stage it as a plain 2D tile (TDM-friendly).
// ---------------------------------------------------------------------------
__global__ __launch_bounds__(256) void qkv_proj_kernel(
    const float* __restrict__ x,
    const _Float16* __restrict__ Wt,   // [3][D][C] f16
    _Float16* __restrict__ out_qkv)    // q | k (row-major) | vT ([b][d][t])
{
  __shared__ _Float16 wbuf[2][128 * 32] __attribute__((aligned(32))); // [d][c_local]

  const int tid  = threadIdx.x;
  const int w    = tid >> 5;
  const int lane = tid & 31;
  const int n    = lane & 15;
  const int h    = lane >> 4;
  const int mat  = blockIdx.y;
  const int rowbase = blockIdx.x * 128 + w * 16;

  const _Float16* Wm = Wt + (size_t)mat * C_ * D_;
  const float* xr    = x + (size_t)(rowbase + n) * C_;

  // staging assignment: thread t copies 16 f16 of Wt chunk (d = t>>1, half = t&1)
  const int sd = tid >> 1;
  const int sh = tid & 1;
  const _Float16* wsrc = Wm + (size_t)sd * C_ + sh * 16;

  v8f acc[8];
  #pragma unroll
  for (int t = 0; t < 8; ++t)
    #pragma unroll
    for (int e = 0; e < 8; ++e) acc[t][e] = 0.0f;

  // prologue: prefetch chunk 0 (weights + x) into registers
  V16 wstage;
  wstage.v = *(const v16h*)(wsrc);
  const float4* xp0 = (const float4*)(xr);
  float4 xa0 = xp0[h * 2 + 0];
  float4 xa1 = xp0[h * 2 + 1];
  float4 xa2 = xp0[4 + h * 2 + 0];
  float4 xa3 = xp0[4 + h * 2 + 1];

  for (int kk = 0; kk < C_ / 32; ++kk) {
    const int buf = kk & 1;
    *(v16h*)(&wbuf[buf][sd * 32 + sh * 16]) = wstage.v;
    __syncthreads();

    v16h a;
    a[0]  = (_Float16)xa0.x; a[1]  = (_Float16)xa0.y; a[2]  = (_Float16)xa0.z; a[3]  = (_Float16)xa0.w;
    a[4]  = (_Float16)xa1.x; a[5]  = (_Float16)xa1.y; a[6]  = (_Float16)xa1.z; a[7]  = (_Float16)xa1.w;
    a[8]  = (_Float16)xa2.x; a[9]  = (_Float16)xa2.y; a[10] = (_Float16)xa2.z; a[11] = (_Float16)xa2.w;
    a[12] = (_Float16)xa3.x; a[13] = (_Float16)xa3.y; a[14] = (_Float16)xa3.z; a[15] = (_Float16)xa3.w;

    if (kk + 1 < C_ / 32) {
      const int c1 = (kk + 1) * 32;
      wstage.v = *(const v16h*)(wsrc + c1);
      const float4* xp = (const float4*)(xr + c1);
      xa0 = xp[h * 2 + 0];
      xa1 = xp[h * 2 + 1];
      xa2 = xp[4 + h * 2 + 0];
      xa3 = xp[4 + h * 2 + 1];
    }

    #pragma unroll
    for (int nt = 0; nt < 8; ++nt) {
      V16 bu;
      bu.u[0] = *(const uint4*)(&wbuf[buf][(nt * 16 + n) * 32 + h * 8]);
      bu.u[1] = *(const uint4*)(&wbuf[buf][(nt * 16 + n) * 32 + 16 + h * 8]);
      const v16h b = bu.v;
      acc[nt] = __builtin_amdgcn_wmma_f32_16x16x32_f16(
          false, a, false, b, (short)0, acc[nt], false, false);
    }
  }

  if (mat < 2) {
    // q / k: row-major f16. D layout: VGPR v holds row v + 8h, col n.
    _Float16* out = out_qkv + (size_t)mat * ((size_t)B_ * T_ * D_);
    #pragma unroll
    for (int nt = 0; nt < 8; ++nt)
      #pragma unroll
      for (int v = 0; v < 8; ++v)
        out[(size_t)(rowbase + v + 8 * h) * D_ + nt * 16 + n] = (_Float16)acc[nt][v];
  } else {
    // v: transposed store vT[b][d][t]
    _Float16* vt = out_qkv + (size_t)2 * B_ * T_ * D_;
    #pragma unroll
    for (int nt = 0; nt < 8; ++nt)
      #pragma unroll
      for (int v = 0; v < 8; ++v) {
        const int row = rowbase + v + 8 * h;        // global token index
        const int bb  = row >> 11;                  // row / T_
        const int t   = row & (T_ - 1);
        vt[((size_t)bb * D_ + nt * 16 + n) * T_ + t] = (_Float16)acc[nt][v];
      }
  }
}

// ---------------------------------------------------------------------------
// Kernel 2: causal flash attention. 4 waves/block = 64 query rows.
// K and vT blocks staged via double-buffered TDM (wave 0 issues next block's
// two descriptors, then s_wait_tensorcnt 2 -> current pair complete while the
// next DMA overlaps compute). Online softmax with 16-lane shfl_xor row
// reductions; P routed through a private LDS tile (D layout -> A layout).
// ---------------------------------------------------------------------------
__global__ __launch_bounds__(128) void flash_attn_kernel(
    const _Float16* __restrict__ qkv, float* __restrict__ out)
{
  const size_t MSZ = (size_t)B_ * T_ * D_;
  const _Float16* qh  = qkv;
  const _Float16* kh  = qkv + MSZ;
  const _Float16* vth = qkv + 2 * MSZ;   // [b][d][t]

  __shared__ _Float16 kbuf[2][32 * 128]  __attribute__((aligned(64))); // [key][d]
  __shared__ _Float16 vbufT[2][128 * 32] __attribute__((aligned(64))); // [d][key]
  __shared__ _Float16 pbuf[4][16 * 32]   __attribute__((aligned(32))); // per wave [row][key]

  const int tid   = threadIdx.x;
  const int w     = tid >> 5;
  const int lane  = tid & 31;
  const int n     = lane & 15;
  const int h     = lane >> 4;
  const int b     = blockIdx.y;
  const int q0    = blockIdx.x * 64;
  const int qbase = q0 + w * 16;
  const int qmax  = qbase + 15;
  const int jend  = q0 + 64;

  const _Float16* qb  = qh  + (size_t)b * T_ * D_;
  const _Float16* kb  = kh  + (size_t)b * T_ * D_;
  const _Float16* vtb = vth + (size_t)b * D_ * T_;

#if HAVE_TDM
  // 2D tile descriptor helper: copy [rows x width] f16 tile (row stride
  // "stride" elements) from gsrc into contiguous LDS at laddr.
  auto tdm_load_2d = [](const _Float16* gsrc, unsigned int laddr,
                        unsigned int width, unsigned int rows,
                        unsigned int stride, unsigned int tensor_rows) {
    const unsigned long long gaddr = (unsigned long long)(uintptr_t)gsrc;
    u32x4 g0;
    g0[0] = 1u;                                        // count=1 (valid user D#)
    g0[1] = laddr;                                     // lds_addr
    g0[2] = (unsigned int)gaddr;                       // global_addr[31:0]
    g0[3] = (unsigned int)(gaddr >> 32) | (2u << 30);  // addr[56:32] | type=2
    i32x8 g1;
    g1[0] = 0x00010000;                                // data_size=1 (2 bytes)
    g1[1] = (int)((stride & 0xFFFFu) << 16);           // tensor_dim0 lo16
    g1[2] = (int)((stride >> 16) | (tensor_rows << 16)); // dim0 hi | dim1 lo16
    g1[3] = (int)((tensor_rows >> 16) | (width << 16)); // dim1 hi | tile_dim0
    g1[4] = (int)rows;                                 // tile_dim1 | tile_dim2=0
    g1[5] = (int)stride;                               // tensor_dim0_stride lo32
    g1[6] = 0;
    g1[7] = 0;
    i32x4 z4; z4[0] = 0; z4[1] = 0; z4[2] = 0; z4[3] = 0;
    i32x8 z8; z8[0] = 0; z8[1] = 0; z8[2] = 0; z8[3] = 0;
              z8[4] = 0; z8[5] = 0; z8[6] = 0; z8[7] = 0;
    __builtin_amdgcn_tensor_load_to_lds(g0, g1, z4, z4, z8, 0);
  };
#endif

  // Q fragments: 4 chunks of 32 over D=128, held in registers
  v16h qf[4];
  {
    const uint4* qrow = (const uint4*)(qb + (size_t)(qbase + n) * D_);
    #pragma unroll
    for (int cc = 0; cc < 4; ++cc) {
      V16 t;
      t.u[0] = qrow[cc * 4 + 2 * h];
      t.u[1] = qrow[cc * 4 + 2 * h + 1];
      qf[cc] = t.v;
    }
  }

  v8f acc[8];
  #pragma unroll
  for (int t = 0; t < 8; ++t)
    #pragma unroll
    for (int e = 0; e < 8; ++e) acc[t][e] = 0.0f;
  float mrow[8], lrow[8];
  #pragma unroll
  for (int v = 0; v < 8; ++v) { mrow[v] = -3.0e38f; lrow[v] = 0.0f; }

  const float scale = 0.08838834764831845f; // 1/sqrt(128)

#if HAVE_TDM
  if (w == 0) {  // prologue: DMA block 0 into buffer 0
    tdm_load_2d(kb,  (unsigned int)(uintptr_t)(&kbuf[0][0]),  128u, 32u, 128u, (unsigned)T_);
    tdm_load_2d(vtb, (unsigned int)(uintptr_t)(&vbufT[0][0]), 32u, 128u, (unsigned)T_, 128u);
  }
#endif

  for (int j0 = 0; j0 < jend; j0 += 32) {
    const int buf = (j0 >> 5) & 1;
#if HAVE_TDM
    if (w == 0) {
      if (j0 + 32 < jend) {
        const int j1 = j0 + 32;
        tdm_load_2d(kb + (size_t)j1 * D_, (unsigned int)(uintptr_t)(&kbuf[buf ^ 1][0]),
                    128u, 32u, 128u, (unsigned)T_);
        tdm_load_2d(vtb + j1, (unsigned int)(uintptr_t)(&vbufT[buf ^ 1][0]),
                    32u, 128u, (unsigned)T_, 128u);
        __builtin_amdgcn_s_wait_tensorcnt(2);  // current pair complete
      } else {
        __builtin_amdgcn_s_wait_tensorcnt(0);
      }
    }
#else
    // ---- manual staging: K block [32][128] (coalesced b128) ----
    {
      const uint4* src = (const uint4*)(kb + (size_t)(j0 + (tid >> 2)) * D_ + (tid & 3) * 32);
      uint4* dst = (uint4*)(&kbuf[buf][(tid >> 2) * 128 + (tid & 3) * 32]);
      #pragma unroll
      for (int i = 0; i < 4; ++i) dst[i] = src[i];
    }
    // ---- manual staging: vT rows are contiguous in t ----
    {
      const uint4* src = (const uint4*)(vtb + (size_t)tid * T_ + j0);
      uint4* dst = (uint4*)(&vbufT[buf][tid * 32]);
      #pragma unroll
      for (int i = 0; i < 4; ++i) dst[i] = src[i];
    }
#endif
    __syncthreads();

    if (j0 <= qmax) {  // wave-uniform: EXEC stays all-ones around WMMA
      // ---- S = Q K^T : two 16x16 score tiles (keys j0.. and j0+16..) ----
      v8f s0, s1;
      #pragma unroll
      for (int e = 0; e < 8; ++e) { s0[e] = 0.0f; s1[e] = 0.0f; }
      #pragma unroll
      for (int cc = 0; cc < 4; ++cc) {
        v16h b0 = *(const v16h*)(&kbuf[buf][n * 128 + cc * 32 + h * 16]);
        v16h b1 = *(const v16h*)(&kbuf[buf][(n + 16) * 128 + cc * 32 + h * 16]);
        s0 = __builtin_amdgcn_wmma_f32_16x16x32_f16(false, qf[cc], false, b0, (short)0, s0, false, false);
        s1 = __builtin_amdgcn_wmma_f32_16x16x32_f16(false, qf[cc], false, b1, (short)0, s1, false, false);
      }

      // ---- scale, causal mask, online softmax (row = v + 8h, col in lanes) ----
      float p0[8], p1[8], alpha[8];
      #pragma unroll
      for (int v = 0; v < 8; ++v) {
        const int qrow = qbase + v + 8 * h;
        float t0 = s0[v] * scale;
        float t1 = s1[v] * scale;
        if (j0 + n > qrow)      t0 = -3.0e38f;
        if (j0 + 16 + n > qrow) t1 = -3.0e38f;
        float bm = fmaxf(t0, t1);
        bm = fmaxf(bm, __shfl_xor(bm, 1));
        bm = fmaxf(bm, __shfl_xor(bm, 2));
        bm = fmaxf(bm, __shfl_xor(bm, 4));
        bm = fmaxf(bm, __shfl_xor(bm, 8));
        const float mnew = fmaxf(mrow[v], bm);
        const float al   = __expf(mrow[v] - mnew);
        const float e0   = __expf(t0 - mnew);
        const float e1   = __expf(t1 - mnew);
        float rs = e0 + e1;
        rs += __shfl_xor(rs, 1);
        rs += __shfl_xor(rs, 2);
        rs += __shfl_xor(rs, 4);
        rs += __shfl_xor(rs, 8);
        lrow[v]  = lrow[v] * al + rs;
        mrow[v]  = mnew;
        alpha[v] = al;
        p0[v] = e0;
        p1[v] = e1;
      }

      // ---- P: D layout -> LDS -> A layout (within-wave, dscnt only) ----
      #pragma unroll
      for (int v = 0; v < 8; ++v) {
        pbuf[w][(v + 8 * h) * 32 + n]      = (_Float16)p0[v];
        pbuf[w][(v + 8 * h) * 32 + 16 + n] = (_Float16)p1[v];
      }
      asm volatile("s_wait_dscnt 0" ::: "memory");
      V16 pa;
      pa.u[0] = *(const uint4*)(&pbuf[w][n * 32 + h * 16]);
      pa.u[1] = *(const uint4*)(&pbuf[w][n * 32 + h * 16 + 8]);
      const v16h pf = pa.v;

      // ---- O = alpha*O + P V ----
      #pragma unroll
      for (int nt = 0; nt < 8; ++nt) {
        v16h vf = *(const v16h*)(&vbufT[buf][(nt * 16 + n) * 32 + h * 16]);
        v8f cacc = acc[nt];
        #pragma unroll
        for (int v = 0; v < 8; ++v) cacc[v] *= alpha[v];
        acc[nt] = __builtin_amdgcn_wmma_f32_16x16x32_f16(
            false, pf, false, vf, (short)0, cacc, false, false);
      }
    }
    __syncthreads();
  }

  // ---- epilogue: O / l ----
  float* ob = out + (size_t)b * T_ * D_;
  #pragma unroll
  for (int v = 0; v < 8; ++v) {
    const float inv = 1.0f / lrow[v];
    const int row = qbase + v + 8 * h;
    #pragma unroll
    for (int nt = 0; nt < 8; ++nt)
      ob[(size_t)row * D_ + nt * 16 + n] = acc[nt][v] * inv;
  }
}

extern "C" void kernel_launch(void* const* d_in, const int* in_sizes, int n_in,
                              void* d_out, int out_size, void* d_ws, size_t ws_size,
                              hipStream_t stream) {
  (void)in_sizes; (void)n_in; (void)out_size; (void)ws_size;
  const float* x  = (const float*)d_in[0];
  const float* Wq = (const float*)d_in[1];
  const float* Wk = (const float*)d_in[2];
  const float* Wv = (const float*)d_in[3];

  _Float16* qkv = (_Float16*)d_ws;                          // q | k | vT, 12 MB
  _Float16* Wt  = qkv + (size_t)3 * B_ * T_ * D_;           // 3 * D*C f16 = 0.75 MB

  wt_convert_kernel<<<(3 * C_ * D_) / 256, 256, 0, stream>>>(Wq, Wk, Wv, Wt);
  qkv_proj_kernel<<<dim3((B_ * T_) / 128, 3), 256, 0, stream>>>(x, Wt, qkv);
  flash_attn_kernel<<<dim3(T_ / 64, B_), 128, 0, stream>>>(qkv, (float*)d_out);
}
